// BahdanauAttention_2972117368951
// MI455X (gfx1250) — matrix operand order repaired and checked by reading, not compiled, not run on hardware
//
#include <hip/hip_runtime.h>
#include <hip/hip_bf16.h>

#define HD 2048
#define BATCH 16
#define SEQ 2048
#define TT 16

typedef __attribute__((ext_vector_type(16))) __bf16 v16bf;
typedef __attribute__((ext_vector_type(8)))  __bf16 v8bf;
typedef __attribute__((ext_vector_type(8)))  float  v8f;

// ---------------- f32 -> bf16 conversion ----------------
__global__ void bah_cvt_bf16(const float* __restrict__ src, __bf16* __restrict__ dst, int n) {
    int i = blockIdx.x * blockDim.x + threadIdx.x;
    int stride = gridDim.x * blockDim.x;
    for (; i < n; i += stride) dst[i] = (__bf16)src[i];
}

// q[b,h] = query[0][b][T-1][h]
__global__ void bah_cvt_q(const float* __restrict__ query, __bf16* __restrict__ qb) {
    int i = blockIdx.x * blockDim.x + threadIdx.x;
    if (i < BATCH * HD) {
        int b = i >> 11, h = i & (HD - 1);
        qb[i] = (__bf16)query[((size_t)(b * TT + (TT - 1))) * HD + h];
    }
}

// ---------------- WMMA fragment loaders from GLOBAL (ISA 7.12.2 layouts) ----------------
// A: 16x32 (MxK). lane L (<16): row L, K {k0..k0+7, k0+16..k0+23}; lane>=16: +8.
__device__ __forceinline__ v16bf load_frag_A(const __bf16* __restrict__ p, int ld,
                                             int row0, int k0, int lane) {
    int r  = row0 + (lane & 15);
    int kk = k0 + ((lane >> 4) << 3);           // 0 or 8
    const __bf16* base = p + (size_t)r * ld + kk;
    v8bf c0 = *(const v8bf*)(base);             // K kk..kk+7   -> elems 0..7
    v8bf c1 = *(const v8bf*)(base + 16);        // K kk+16..+23 -> elems 8..15
    v16bf out;
#pragma unroll
    for (int i = 0; i < 8; ++i) { out[i] = c0[i]; out[i + 8] = c1[i]; }
    return out;
}

// B: 32x16 (KxN), with B[k][n] = W[n0+n][k] (W row-major, ld).
__device__ __forceinline__ v16bf load_frag_B(const __bf16* __restrict__ p, int ld,
                                             int n0, int k0, int lane) {
    int r  = n0 + (lane & 15);
    int kk = k0 + ((lane >> 4) << 3);           // 0 or 8
    const __bf16* base = p + (size_t)r * ld + kk;
    v8bf c0 = *(const v8bf*)(base);             // K kk..kk+7  -> elems 0..7
    v8bf c1 = *(const v8bf*)(base + 16);        // K kk+16..+23 -> elems 8..15
    v16bf out;
#pragma unroll
    for (int i = 0; i < 8; ++i) { out[i] = c0[i]; out[i + 8] = c1[i]; }
    return out;
}

// ---------------- small GEMM (q_proj, M=16): one 16x16 tile per wave ----------------
__global__ void bah_gemm_bf16_nt(const __bf16* __restrict__ A, const __bf16* __restrict__ W,
                                 const float* __restrict__ bias, float* __restrict__ C,
                                 int n_tiles) {
    int wave = (blockIdx.x * blockDim.x + threadIdx.x) >> 5;
    int lane = threadIdx.x & 31;
    int m0 = (wave / n_tiles) << 4;
    int n0 = (wave % n_tiles) << 4;

    v8f acc = {};
#pragma unroll 4
    for (int k0 = 0; k0 < HD; k0 += 32) {
        v16bf a = load_frag_A(A, HD, m0, k0, lane);
        v16bf b = load_frag_B(W, HD, n0, k0, lane);
        acc = __builtin_amdgcn_wmma_f32_16x16x32_bf16(false, a, false, b,
                                                      (short)0, acc, false, false);
    }
    int mbase = m0 + ((lane >> 4) << 3);
    int n     = n0 + (lane & 15);
    float bv  = bias[n];
#pragma unroll
    for (int r = 0; r < 8; ++r)
        C[(size_t)(mbase + r) * HD + n] = acc[r] + bv;
}

// ---------------- async global->LDS helpers (gfx1250, ASYNCcnt-tracked) ----------------
// vdst VGPR holds the per-lane LDS byte address; addr VGPR pair holds the global address.
__device__ __forceinline__ void async_g2l_b128(unsigned lds_byte_addr, const void* gaddr) {
    asm volatile("global_load_async_to_lds_b128 %0, %1, off"
                 :: "v"(lds_byte_addr), "v"(gaddr) : "memory");
}
__device__ __forceinline__ void wait_async0() {
    asm volatile("s_wait_asynccnt 0" ::: "memory");
}
// flat pointer to a __shared__ object: LDS byte offset lives in the low 32 bits
__device__ __forceinline__ unsigned lds_addr_of(const void* p) {
    return (unsigned)(unsigned long long)p;
}

// ---------------- tiled GEMM (v_proj): 128x128 macro-tile per block ----------------
// C[m,n] = sum_k A[m,k]*W[n,k] + bias[n].  Block = 256 thr = 8 waves (2 M x 4 N),
// wave = 64x32 sub-tile = 4x2 WMMA accumulators. Double-buffered LDS panels filled
// by GLOBAL_LOAD_ASYNC_TO_LDS_B128 (no VGPR round-trip), rows padded to 40 bf16.
#define BK 32
#define LRS 40
#define PANEL_BYTES (128 * LRS * 2)   // one buffer of As or Bs

__global__ void __launch_bounds__(256)
bah_gemm_tiled(const __bf16* __restrict__ A, const __bf16* __restrict__ W,
               const float* __restrict__ bias, float* __restrict__ C) {
    const int m0 = blockIdx.y << 7;
    const int n0 = blockIdx.x << 7;
    const int t = threadIdx.x;
    const int wave = t >> 5, lane = t & 31;
    const int wm = (wave & 1) << 6;   // 0 or 64
    const int wn = (wave >> 1) << 5;  // 0,32,64,96

    __shared__ __align__(16) __bf16 As[2][128 * LRS];
    __shared__ __align__(16) __bf16 Bs[2][128 * LRS];

    // staging: each thread owns one 16-bf16 (32 B) chunk of each panel per k-step
    const int lrow = t >> 1;           // 0..127
    const int lcol = (t & 1) << 4;     // 0 or 16
    const __bf16* gA = A + (size_t)(m0 + lrow) * HD + lcol;
    const __bf16* gB = W + (size_t)(n0 + lrow) * HD + lcol;
    const unsigned ldsA0 = lds_addr_of(&As[0][lrow * LRS + lcol]);
    const unsigned ldsB0 = lds_addr_of(&Bs[0][lrow * LRS + lcol]);

    // prologue: panel k=0 -> buffer 0 (async)
    async_g2l_b128(ldsA0,      gA);
    async_g2l_b128(ldsA0 + 16, gA + 8);
    async_g2l_b128(ldsB0,      gB);
    async_g2l_b128(ldsB0 + 16, gB + 8);

    v8f acc[4][2];
#pragma unroll
    for (int i = 0; i < 4; ++i)
#pragma unroll
        for (int j = 0; j < 2; ++j) acc[i][j] = (v8f){};

    const int NK = HD / BK;  // 64
    for (int kt = 0; kt < NK; ++kt) {
        const int cur = kt & 1;
        wait_async0();        // my copies of panel `cur` have landed in LDS
        __syncthreads();      // everyone's landed; everyone done reading buffer `nxt`

        if (kt + 1 < NK) {    // overlap next panel's async copy with compute below
            const int nxt = cur ^ 1;
            const unsigned dA = ldsA0 + nxt * PANEL_BYTES;
            const unsigned dB = ldsB0 + nxt * PANEL_BYTES;
            const __bf16* pA = gA + (size_t)(kt + 1) * BK;
            const __bf16* pB = gB + (size_t)(kt + 1) * BK;
            async_g2l_b128(dA,      pA);
            async_g2l_b128(dA + 16, pA + 8);
            async_g2l_b128(dB,      pB);
            async_g2l_b128(dB + 16, pB + 8);
        }

        // fragment loads from LDS
        v16bf af[4], bfr[2];
        {
            const int kkA = (lane >> 4) << 3;        // 0 or 8
            const int rA  = wm + (lane & 15);
#pragma unroll
            for (int i = 0; i < 4; ++i) {
                const __bf16* p = &As[cur][(rA + (i << 4)) * LRS + kkA];
                v8bf c0 = *(const v8bf*)(p);
                v8bf c1 = *(const v8bf*)(p + 16);
#pragma unroll
                for (int q = 0; q < 8; ++q) { af[i][q] = c0[q]; af[i][q + 8] = c1[q]; }
            }
            const int kkB = (lane >> 4) << 3;        // 0 or 8
            const int rB  = wn + (lane & 15);
#pragma unroll
            for (int j = 0; j < 2; ++j) {
                const __bf16* p = &Bs[cur][(rB + (j << 4)) * LRS + kkB];
                v8bf c0 = *(const v8bf*)(p);
                v8bf c1 = *(const v8bf*)(p + 16);
#pragma unroll
                for (int q = 0; q < 8; ++q) { bfr[j][q] = c0[q]; bfr[j][q + 8] = c1[q]; }
            }
        }
#pragma unroll
        for (int i = 0; i < 4; ++i)
#pragma unroll
            for (int j = 0; j < 2; ++j)
                acc[i][j] = __builtin_amdgcn_wmma_f32_16x16x32_bf16(
                    false, af[i], false, bfr[j], (short)0, acc[i][j], false, false);
    }

    // store 4x2 tiles
    const int mb = m0 + wm + ((lane >> 4) << 3);
    const int nb = n0 + wn + (lane & 15);
#pragma unroll
    for (int j = 0; j < 2; ++j) {
        const int n = nb + (j << 4);
        const float bv = bias[n];
#pragma unroll
        for (int i = 0; i < 4; ++i)
#pragma unroll
            for (int r = 0; r < 8; ++r)
                C[(size_t)(mb + (i << 4) + r) * HD + n] = acc[i][j][r] + bv;
    }
}

// ---------------- scores[b][s] = tanh(Qp[b,:] + Vp[s,:]) . Vw + Vb ----------------
#define SB 4
__global__ void bah_scores(const float* __restrict__ Qp, const float* __restrict__ Vp,
                           const float* __restrict__ Vw, const float* __restrict__ Vb_s,
                           float* __restrict__ scores) {
    int s0 = blockIdx.x * SB;
    int t = threadIdx.x;
    float vw[8], vp[SB][8];
#pragma unroll
    for (int j = 0; j < 8; ++j) vw[j] = Vw[t + (j << 8)];
#pragma unroll
    for (int s = 0; s < SB; ++s)
#pragma unroll
        for (int j = 0; j < 8; ++j)
            vp[s][j] = Vp[(size_t)(s0 + s) * HD + t + (j << 8)];

    float partial[SB][BATCH];
#pragma unroll
    for (int b = 0; b < BATCH; ++b) {
        float qp[8];
#pragma unroll
        for (int j = 0; j < 8; ++j) qp[j] = Qp[(size_t)b * HD + t + (j << 8)];
#pragma unroll
        for (int s = 0; s < SB; ++s) {
            float acc = 0.f;
#pragma unroll
            for (int j = 0; j < 8; ++j)
                acc += tanhf(qp[j] + vp[s][j]) * vw[j];
            partial[s][b] = acc;
        }
    }

    __shared__ float red[8][SB][BATCH];
    int wave = t >> 5, lane = t & 31;
#pragma unroll
    for (int s = 0; s < SB; ++s)
#pragma unroll
        for (int b = 0; b < BATCH; ++b) {
            float v = partial[s][b];
            for (int off = 16; off > 0; off >>= 1) v += __shfl_xor(v, off, 32);
            if (lane == 0) red[wave][s][b] = v;
        }
    __syncthreads();
    if (t < SB * BATCH) {
        int s = t >> 4, b = t & 15;
        float sum = 0.f;
#pragma unroll
        for (int w = 0; w < 8; ++w) sum += red[w][s][b];
        scores[(size_t)b * SEQ + s0 + s] = sum + Vb_s[0];
    }
}

// ---------------- softmax over S per batch ----------------
__global__ void bah_softmax(const float* __restrict__ scores, float* __restrict__ alphas) {
    int b = blockIdx.x;
    int t = threadIdx.x;
    __shared__ float red[8];
    float v[8];
#pragma unroll
    for (int j = 0; j < 8; ++j) v[j] = scores[(size_t)b * SEQ + t + (j << 8)];
    float m = v[0];
#pragma unroll
    for (int j = 1; j < 8; ++j) m = fmaxf(m, v[j]);
    for (int off = 16; off > 0; off >>= 1) m = fmaxf(m, __shfl_xor(m, off, 32));
    if ((t & 31) == 0) red[t >> 5] = m;
    __syncthreads();
    float m0 = red[0];
#pragma unroll
    for (int w = 1; w < 8; ++w) m0 = fmaxf(m0, red[w]);
    __syncthreads();
    float sum = 0.f;
#pragma unroll
    for (int j = 0; j < 8; ++j) { v[j] = expf(v[j] - m0); sum += v[j]; }
    for (int off = 16; off > 0; off >>= 1) sum += __shfl_xor(sum, off, 32);
    if ((t & 31) == 0) red[t >> 5] = sum;
    __syncthreads();
    float tot = 0.f;
#pragma unroll
    for (int w = 0; w < 8; ++w) tot += red[w];
    float inv = 1.0f / tot;
#pragma unroll
    for (int j = 0; j < 8; ++j)
        alphas[(size_t)b * SEQ + t + (j << 8)] = v[j] * inv;
}

// ---------------- context[b,h] = sum_s alpha[b,s] * values[b,s,h] ----------------
__global__ void bah_context(const float* __restrict__ values, const float* __restrict__ alphas,
                            float* __restrict__ context) {
    int b  = blockIdx.x >> 3;
    int h  = ((blockIdx.x & 7) << 8) + threadIdx.x;
    __shared__ float al[SEQ];
#pragma unroll
    for (int j = 0; j < 8; ++j) {
        int s = threadIdx.x + (j << 8);
        al[s] = alphas[(size_t)b * SEQ + s];
    }
    __syncthreads();
    const float* vb = values + (size_t)b * SEQ * HD + h;
    float acc = 0.f;
#pragma unroll 4
    for (int s = 0; s < SEQ; ++s)
        acc = fmaf(al[s], vb[(size_t)s * HD], acc);
    context[(size_t)b * HD + h] = acc;
}

// ---------------- host-side orchestration ----------------
extern "C" void kernel_launch(void* const* d_in, const int* in_sizes, int n_in,
                              void* d_out, int out_size, void* d_ws, size_t ws_size,
                              hipStream_t stream) {
    (void)in_sizes; (void)n_in; (void)out_size; (void)ws_size;
    const float* query  = (const float*)d_in[0];
    const float* values = (const float*)d_in[1];
    // d_in[2] = mask (unused by reference math)
    const float* W1w = (const float*)d_in[3];
    const float* W1b = (const float*)d_in[4];
    const float* W2w = (const float*)d_in[5];
    const float* W2b = (const float*)d_in[6];
    const float* Vw  = (const float*)d_in[7];
    const float* Vb  = (const float*)d_in[8];

    char* ws = (char*)d_ws;
    __bf16* Vbf  = (__bf16*)ws;                 ws += (size_t)SEQ * HD * 2;     // 8 MB
    __bf16* W1bf = (__bf16*)ws;                 ws += (size_t)HD * HD * 2;      // 8 MB
    __bf16* W2bf = (__bf16*)ws;                 ws += (size_t)HD * HD * 2;      // 8 MB
    __bf16* Qbf  = (__bf16*)ws;                 ws += (size_t)BATCH * HD * 2;   // 64 KB
    float*  Qp   = (float*)ws;                  ws += (size_t)BATCH * HD * 4;   // 128 KB
    float*  Vp   = (float*)ws;                  ws += (size_t)SEQ * HD * 4;     // 16 MB
    float*  Sc   = (float*)ws;                  /* BATCH*SEQ*4 = 128 KB */

    float* ctx_out   = (float*)d_out;                 // (B,1,H)
    float* alpha_out = ctx_out + (size_t)BATCH * HD;  // (B,1,S)

    const int nHH = HD * HD;
    bah_cvt_bf16<<<1024, 256, 0, stream>>>(values, Vbf, SEQ * HD); // values[0]
    bah_cvt_bf16<<<1024, 256, 0, stream>>>(W1w, W1bf, nHH);
    bah_cvt_bf16<<<1024, 256, 0, stream>>>(W2w, W2bf, nHH);
    bah_cvt_q  <<<(BATCH * HD) / 256, 256, 0, stream>>>(query, Qbf);

    // q_proj: 1 m-tile x 128 n-tiles = 128 waves
    bah_gemm_bf16_nt<<<16, 256, 0, stream>>>(Qbf, W1bf, W1b, Qp, HD / 16);
    // v_proj: 2048x2048 via 128x128 macro-tiles -> 16x16 blocks
    {
        dim3 grid(HD / 128, SEQ / 128);
        bah_gemm_tiled<<<grid, 256, 0, stream>>>(Vbf, W2bf, W2b, Vp);
    }

    bah_scores<<<SEQ / SB, 256, 0, stream>>>(Qp, Vp, Vw, Vb, Sc);
    bah_softmax<<<BATCH, 256, 0, stream>>>(Sc, alpha_out);
    bah_context<<<BATCH * (HD / 256), 256, 0, stream>>>(values, alpha_out, ctx_out);
}
